// ResRnn_44032004718801
// MI455X (gfx1250) — compile-verified, matrix operand order checked by reading, not run on hardware
//
#include <hip/hip_runtime.h>

// ---------------------------------------------------------------------------
// ResRnn on MI455X (gfx1250): per-step bf16 WMMA GEMMs, fp32 accumulate,
// LDS double-buffered K-pipeline fed by CDNA5 async global->LDS copies.
//   s_t   = [x_t | s_{t-1}[:, :1984]]
//   h     = relu(s @ W1^T + b1)
//   s'    = 0.9*s + 0.1*(h @ W2^T + b2)
// Output: final state[:, 1024:2048]  (512 x 1024 fp32)
// ---------------------------------------------------------------------------

typedef __bf16 bf16;
typedef __attribute__((ext_vector_type(16))) __bf16 v16bf;
typedef __attribute__((ext_vector_type(8)))  float  v8f;

#define SEQ     256
#define BATCH   512
#define IN_W    64
#define STREAMW 2048
#define OUT_W   1024
#define KEEP    (STREAMW - IN_W)   // 1984
#define LINC    0.9f

#define BK      64                 // K-tile depth
#define LDP     72                 // padded LDS row (halves): banks spaced by 4

__device__ __forceinline__ unsigned short f32_to_bf16_bits(float f) {
    union { float f; unsigned u; } v; v.f = f;
    unsigned u = v.u;
    if ((u & 0x7fffffffu) > 0x7f800000u)          // NaN: quiet it
        return (unsigned short)((u >> 16) | 0x40);
    u += 0x7fffu + ((u >> 16) & 1u);              // round-to-nearest-even
    return (unsigned short)(u >> 16);
}

// Async global -> LDS copy, 16B per lane; tracked by ASYNCcnt.
// Low 32 bits of a generic LDS pointer are the LDS byte offset (aperture
// layout: addr[31:0] = LDS offset), which is what VDST expects.
__device__ __forceinline__ void async_load_b128(unsigned lds_off,
                                                const void* gptr) {
    asm volatile("global_load_async_to_lds_b128 %0, %1, off"
                 :: "v"(lds_off), "v"(gptr) : "memory");
}

template <int N>
__device__ __forceinline__ void wait_asynccnt() {
    asm volatile("s_wait_asynccnt %0" :: "i"(N) : "memory");
}

__device__ __forceinline__ v16bf load_frag_lds(const unsigned short* p) {
    v16bf v;
    __builtin_memcpy(&v, p, 32);   // 2x ds_load_b128 (16B-aligned)
    return v;
}

// ---------------------------------------------------------------------------
// Shared GEMM core: acc(128x128 block tile) += A(128xK) * W(128 rows x K)^T.
// 8 waves as 4(m) x 2(n); wave tile 32x64 = 2x4 WMMA 16x16 tiles.
// A and B fragments have identical per-lane layouts from row-major sources:
// lane L reads 16 contiguous bf16 at row (L&15), k-offset (L>>4)*16.
// ---------------------------------------------------------------------------
__device__ __forceinline__ void gemm_core(
    const unsigned short* __restrict__ Ag,   // [*, STREAMW] row-major
    const unsigned short* __restrict__ Wg,   // [STREAMW, STREAMW] row-major
    int m0blk, int n0blk, v8f acc[2][4]) {
    __shared__ unsigned short lsA[2][128][LDP];
    __shared__ unsigned short lsB[2][128][LDP];

    const int tid  = threadIdx.x;
    const int lane = tid & 31;
    const int wave = tid >> 5;
    const int wm = wave & 3, wn = wave >> 2;
    const int r  = lane & 15;
    const int ko = (lane >> 4) << 4;

    const unsigned short* gA = Ag + (size_t)m0blk * STREAMW;
    const unsigned short* gB = Wg + (size_t)n0blk * STREAMW;

    // Loader: 128 rows x 64 halves (128B) per tile; 8x16B chunks per row;
    // thread -> (row = tid>>3 + 32*i, seg = tid&7): consecutive threads cover
    // one 128B row segment-contiguously (coalesced), 4 chunks x {A,B} each.
    const int lrow = tid >> 3;
    const int lcol = (tid & 7) * 8;   // halves

    auto issue_tile = [&](int buf, int k0) {
#pragma unroll
        for (int i = 0; i < 4; ++i) {
            int row = lrow + i * 32;
            async_load_b128((unsigned)(uintptr_t)&lsA[buf][row][lcol],
                            gA + (size_t)row * STREAMW + k0 + lcol);
            async_load_b128((unsigned)(uintptr_t)&lsB[buf][row][lcol],
                            gB + (size_t)row * STREAMW + k0 + lcol);
        }
    };

    issue_tile(0, 0);                        // 8 async ops outstanding
    const int NT = STREAMW / BK;             // 32 K-tiles
    for (int kt = 0; kt < NT; ++kt) {
        const int buf = kt & 1;
        if (kt + 1 < NT) {
            issue_tile(buf ^ 1, (kt + 1) * BK);  // prefetch next tile
            wait_asynccnt<8>();              // current tile's 8 ops done
        } else {
            wait_asynccnt<0>();
        }
        __syncthreads();                     // tile visible to all waves

#pragma unroll
        for (int kk = 0; kk < BK; kk += 32) {
            v16bf a0 = load_frag_lds(&lsA[buf][wm * 32 + r     ][kk + ko]);
            v16bf a1 = load_frag_lds(&lsA[buf][wm * 32 + 16 + r][kk + ko]);
#pragma unroll
            for (int j = 0; j < 4; ++j) {
                v16bf b = load_frag_lds(&lsB[buf][wn * 64 + j * 16 + r][kk + ko]);
                acc[0][j] = __builtin_amdgcn_wmma_f32_16x16x32_bf16(
                    false, a0, false, b, (short)0, acc[0][j], false, false);
                acc[1][j] = __builtin_amdgcn_wmma_f32_16x16x32_bf16(
                    false, a1, false, b, (short)0, acc[1][j], false, false);
            }
        }
        __syncthreads();                     // reads done before buf reuse
    }
}

// --- GEMM1: H = relu(S @ W1^T + b1) ----------------------------------------
__global__ __launch_bounds__(256) void gemm1_relu(
    const unsigned short* __restrict__ S, const unsigned short* __restrict__ W1,
    const float* __restrict__ b1, unsigned short* __restrict__ H) {
    v8f acc[2][4];
#pragma unroll
    for (int i = 0; i < 2; ++i)
#pragma unroll
        for (int j = 0; j < 4; ++j) acc[i][j] = (v8f){0,0,0,0,0,0,0,0};

    gemm_core(S, W1, blockIdx.y * 128, blockIdx.x * 128, acc);

    const int lane = threadIdx.x & 31;
    const int wave = threadIdx.x >> 5;
    const int m0 = blockIdx.y * 128 + (wave & 3) * 32;
    const int n0 = blockIdx.x * 128 + (wave >> 2) * 64;
    const int nlane = lane & 15;
    const int mo    = (lane >> 4) * 8;    // D: VGPR v -> M=v+8*(lane>=16), N=lane&15
#pragma unroll
    for (int j = 0; j < 4; ++j) {
        int col = n0 + j * 16 + nlane;
        float bias = b1[col];
#pragma unroll
        for (int i = 0; i < 2; ++i) {
            int mbase = m0 + i * 16 + mo;
#pragma unroll
            for (int v = 0; v < 8; ++v) {
                float val = fmaxf(acc[i][j][v] + bias, 0.0f);
                H[(size_t)(mbase + v) * STREAMW + col] = f32_to_bf16_bits(val);
            }
        }
    }
}

// --- GEMM2: out = 0.9*S + 0.1*(H @ W2^T + b2), fused shift/output ----------
__global__ __launch_bounds__(256) void gemm2_mix(
    const unsigned short* __restrict__ H, const unsigned short* __restrict__ W2,
    const float* __restrict__ b2, const float* __restrict__ Scur,
    float* __restrict__ Snf, unsigned short* __restrict__ Snb,
    float* __restrict__ out, int last) {
    v8f acc[2][4];
#pragma unroll
    for (int i = 0; i < 2; ++i)
#pragma unroll
        for (int j = 0; j < 4; ++j) acc[i][j] = (v8f){0,0,0,0,0,0,0,0};

    gemm_core(H, W2, blockIdx.y * 128, blockIdx.x * 128, acc);

    const int lane = threadIdx.x & 31;
    const int wave = threadIdx.x >> 5;
    const int m0 = blockIdx.y * 128 + (wave & 3) * 32;
    const int n0 = blockIdx.x * 128 + (wave >> 2) * 64;
    const int nlane = lane & 15;
    const int mo    = (lane >> 4) * 8;
#pragma unroll
    for (int j = 0; j < 4; ++j) {
        int col = n0 + j * 16 + nlane;
        float bias = b2[col];
#pragma unroll
        for (int i = 0; i < 2; ++i) {
            int mbase = m0 + i * 16 + mo;
#pragma unroll
            for (int v = 0; v < 8; ++v) {
                int row = mbase + v;
                float val = LINC * Scur[(size_t)row * STREAMW + col]
                          + (1.0f - LINC) * (acc[i][j][v] + bias);
                if (!last) {
                    if (col < KEEP) {
                        size_t idx = (size_t)row * STREAMW + col + IN_W;
                        Snf[idx] = val;
                        Snb[idx] = f32_to_bf16_bits(val);
                    }
                } else {
                    if (col >= STREAMW - OUT_W)
                        out[(size_t)row * OUT_W + (col - (STREAMW - OUT_W))] = val;
                }
            }
        }
    }
}

// --- one-time fp32 -> bf16 weight conversion (both matrices) ---------------
__global__ __launch_bounds__(256) void convert_weights(
    const float* __restrict__ W1, const float* __restrict__ W2,
    unsigned short* __restrict__ W1b, unsigned short* __restrict__ W2b, int n) {
    int i = blockIdx.x * 256 + threadIdx.x;
    if (i < n) {
        W1b[i] = f32_to_bf16_bits(W1[i]);
        W2b[i] = f32_to_bf16_bits(W2[i]);
    }
}

// --- zero initial state (fp32 + bf16 copies) --------------------------------
__global__ __launch_bounds__(256) void zero_state(
    float* __restrict__ sf, unsigned short* __restrict__ sb, int n) {
    int i = blockIdx.x * 256 + threadIdx.x;
    if (i < n) { sf[i] = 0.0f; sb[i] = 0; }
}

// --- inject x[t] into columns [0,64) of the current state -------------------
__global__ __launch_bounds__(256) void inject_x(
    const float* __restrict__ xt, float* __restrict__ sf,
    unsigned short* __restrict__ sb) {
    int i = blockIdx.x * 256 + threadIdx.x;      // 512*64 threads
    int b = i >> 6, c = i & 63;
    float v = xt[i];
    sf[b * STREAMW + c] = v;
    sb[b * STREAMW + c] = f32_to_bf16_bits(v);
}

// ---------------------------------------------------------------------------
extern "C" void kernel_launch(void* const* d_in, const int* in_sizes, int n_in,
                              void* d_out, int out_size, void* d_ws, size_t ws_size,
                              hipStream_t stream) {
    const float* x  = (const float*)d_in[0];   // (256, 512, 64)
    const float* W1 = (const float*)d_in[1];   // (2048, 2048)
    const float* b1 = (const float*)d_in[2];
    const float* W2 = (const float*)d_in[3];
    const float* b2 = (const float*)d_in[4];
    float* out = (float*)d_out;                // (512, 1024)

    size_t off = 0;
    auto carve = [&](size_t bytes) {
        void* p = (char*)d_ws + off;
        off += (bytes + 255) & ~(size_t)255;
        return p;
    };
    const size_t WN = (size_t)STREAMW * STREAMW;       // 4M elems
    const size_t SN = (size_t)BATCH * STREAMW;         // 1M elems
    unsigned short* W1b = (unsigned short*)carve(WN * 2);
    unsigned short* W2b = (unsigned short*)carve(WN * 2);
    float*          sf[2]; unsigned short* sb[2];
    sf[0] = (float*)carve(SN * 4);  sb[0] = (unsigned short*)carve(SN * 2);
    sf[1] = (float*)carve(SN * 4);  sb[1] = (unsigned short*)carve(SN * 2);
    unsigned short* Hb = (unsigned short*)carve(SN * 2);

    convert_weights<<<(int)((WN + 255) / 256), 256, 0, stream>>>(
        W1, W2, W1b, W2b, (int)WN);
    zero_state<<<(int)((SN + 255) / 256), 256, 0, stream>>>(
        sf[0], sb[0], (int)SN);

    dim3 gGemm(STREAMW / 128, BATCH / 128);   // (16, 4)
    int cur = 0;
    for (int t = 0; t < SEQ; ++t) {
        int nxt = cur ^ 1;
        inject_x<<<(BATCH * IN_W) / 256, 256, 0, stream>>>(
            x + (size_t)t * BATCH * IN_W, sf[cur], sb[cur]);
        gemm1_relu<<<gGemm, 256, 0, stream>>>(sb[cur], W1b, b1, Hb);
        gemm2_mix<<<gGemm, 256, 0, stream>>>(Hb, W2b, b2, sf[cur],
                                             sf[nxt], sb[nxt], out,
                                             (t == SEQ - 1) ? 1 : 0);
        cur = nxt;
    }
}
